// CausalAttention_12429635355391
// MI455X (gfx1250) — matrix operand order repaired and checked by reading, not compiled, not run on hardware
//
#include <hip/hip_runtime.h>
#include <hip/hip_bf16.h>

// ---------------------------------------------------------------------------
// Causal attention, B=4, T=2048, D=1024 (fp32 in/out).
// Strategy: bf16 WMMA (v_wmma_f32_16x16x32_bf16) for all three matmul stages,
// fp32 softmax/accumulation. Everything fits in the 192MB L2 -> compute bound.
// Workspace usage: 70 MB (xb 16MB | Wt 6MB | Q 16MB | K 16MB | Vt 16MB).
// ---------------------------------------------------------------------------

typedef __bf16 bf16;
typedef __attribute__((ext_vector_type(16))) __bf16 v16bf;
typedef __attribute__((ext_vector_type(8)))  __bf16 v8bf;
typedef __attribute__((ext_vector_type(8)))  float  v8f;

#define BATCH 4
#define TSEQ  2048
#define DDIM  1024

__device__ __forceinline__ v8f zero8() {
  v8f z;
  #pragma unroll
  for (int i = 0; i < 8; ++i) z[i] = 0.0f;
  return z;
}

__device__ __forceinline__ v8f wmma_bf16(v16bf a, v16bf b, v8f c) {
  // D = A(16x32 bf16) * B(32x16 bf16) + C(16x16 f32)
  return __builtin_amdgcn_wmma_f32_16x16x32_bf16(
      /*neg_a=*/false, a, /*neg_b=*/false, b,
      /*c_mod=*/(short)0, c, /*reuse_a=*/false, /*reuse_b=*/false);
}

// A-fragment: 16x32 bf16 tile, row-major source. Lane L holds row (L&15);
// elements 0-15 -> K = (L>>4)*16 + 0..15.
// `rowptr` must point at element [row][0] of the 32-wide K slice.
__device__ __forceinline__ v16bf load_a_frag(const bf16* rowptr, int lane) {
  const int g = (lane >> 4) & 1;
  union { v16bf v; v8bf h[2]; } u;
  u.h[0] = *(const v8bf*)(rowptr + g * 16);
  u.h[1] = *(const v8bf*)(rowptr + g * 16 + 8);
  return u.v;
}

// ---------------------------------------------------------------------------
// Kernel 1: fp32 -> bf16 elementwise convert
// ---------------------------------------------------------------------------
__global__ void cvt_bf16_kernel(const float* __restrict__ src,
                                bf16* __restrict__ dst, int n) {
  int i = blockIdx.x * blockDim.x + threadIdx.x;
  if (i < n) dst[i] = (bf16)src[i];
}

// ---------------------------------------------------------------------------
// Kernel 2: W [K=1024][N=1024] fp32 -> Wt [N][K] bf16 (coalesced writes)
// ---------------------------------------------------------------------------
__global__ void transpose_w_kernel(const float* __restrict__ src,
                                   bf16* __restrict__ dst) {
  int i = blockIdx.x * blockDim.x + threadIdx.x;  // dst index = n*1024 + k
  int n = i >> 10;
  int k = i & 1023;
  dst[i] = (bf16)src[k * 1024 + n];
}

// ---------------------------------------------------------------------------
// Kernel 3: QKV projection GEMM. X[M=8192][1024] (bf16) * W -> Y (bf16).
// One wave computes a 16x64 output tile (4 v8f accumulators).
// blockIdx.z = projection: 0->Q (row major), 1->K (row major), 2->V stored
// transposed as Vt[B][D][T] (contiguous 16B stores straight from C layout).
// ---------------------------------------------------------------------------
__global__ __launch_bounds__(256)
void qkv_gemm_kernel(const bf16* __restrict__ X, const bf16* __restrict__ Wt,
                     bf16* __restrict__ Q, bf16* __restrict__ Kmat,
                     bf16* __restrict__ Vt) {
  constexpr int Kd = DDIM, Nd = DDIM;
  const int lane = threadIdx.x & 31;
  const int wv   = threadIdx.x >> 5;          // 0..7
  const int row  = lane & 15;
  const int g    = lane >> 4;
  const int proj = blockIdx.z;
  const int m0   = (blockIdx.x * 8 + wv) * 16;
  const int n0   = blockIdx.y * 64;

  const bf16* wt = Wt + (size_t)proj * Kd * Nd;      // [N][K] bf16
  const bf16* xr = X + (size_t)(m0 + row) * Kd;      // this lane's A row

  v8f acc[4];
  #pragma unroll
  for (int t = 0; t < 4; ++t) acc[t] = zero8();

  for (int kk = 0; kk < Kd; kk += 32) {
    v16bf a = load_a_frag(xr + kk, lane);
    #pragma unroll
    for (int t = 0; t < 4; ++t) {
      // B frag: column n = row, K contiguous from Wt[n][*]
      const bf16* wr = wt + (size_t)(n0 + t * 16 + row) * Kd + kk + g * 16;
      v16bf bv = *(const v16bf*)wr;
      acc[t] = wmma_bf16(a, bv, acc[t]);
    }
  }

  if (proj < 2) {
    bf16* Y = (proj == 0) ? Q : Kmat;
    #pragma unroll
    for (int t = 0; t < 4; ++t) {
      const int col = n0 + t * 16 + row;
      #pragma unroll
      for (int r = 0; r < 8; ++r) {
        const int m = m0 + g * 8 + r;      // C layout: M = (lane>=16?8:0)+vgpr
        Y[(size_t)m * Nd + col] = (bf16)acc[t][r];
      }
    }
  } else {
    // V transposed: Vt[b][d][t]; rows of C are contiguous t positions.
    const int mbase = m0 + g * 8;
    const int b     = mbase >> 11;          // / TSEQ
    const int tpos  = mbase & (TSEQ - 1);
    #pragma unroll
    for (int t = 0; t < 4; ++t) {
      const int col = n0 + t * 16 + row;    // d index
      union { v8bf v; bf16 e[8]; } u;
      #pragma unroll
      for (int r = 0; r < 8; ++r) u.e[r] = (bf16)acc[t][r];
      *(v8bf*)(Vt + ((size_t)b * DDIM + col) * TSEQ + tpos) = u.v;
    }
  }
}

// ---------------------------------------------------------------------------
// Kernel 4: flash-style causal attention.
// Block = (batch b, 16-query block qb). 8 waves; wave w owns d-slice
// [128w, 128w+128) of the output (8 v8f accumulators).
// Per 32-key block: each wave computes partial S over its 128-d slice
// (8 WMMAs), deterministic LDS tree-reduction, per-row online softmax,
// accumulator rescale, then 8 WMMAs of P*V against Vt.
// ---------------------------------------------------------------------------
__global__ __launch_bounds__(256)
void attn_kernel(const bf16* __restrict__ Q, const bf16* __restrict__ Kmat,
                 const bf16* __restrict__ Vt, float* __restrict__ Out) {
  const int lane = threadIdx.x & 31;
  const int wv   = threadIdx.x >> 5;      // 0..7
  const int row  = lane & 15;             // this lane's softmax row
  const int g    = lane >> 4;             // K-group within fragments
  const int qb   = blockIdx.x;
  const int b    = blockIdx.y;
  const int q0   = qb * 16;
  const int d0   = wv * 128;

  __shared__ float S_part[8][16][32];     // per-wave partial scores (16 KB)
  __shared__ float S_lds[16][32];         // reduced scores (2 KB)
  __shared__ float bc[8][16];             // per-wave row-broadcast scratch

  v8f acc[8];
  #pragma unroll
  for (int t = 0; t < 8; ++t) acc[t] = zero8();
  float m_run = -3.0e38f;
  float l_run = 0.0f;
  // softmax done in exp2 domain: s' = s * (1/sqrt(D)) * log2(e)
  const float c = 1.44269504088896f / 32.0f;

  const bf16* Qrow = Q + ((size_t)b * TSEQ + (q0 + row)) * DDIM;
  const int nkb = (q0 + 16 + 31) >> 5;    // 32-key blocks needed (causal)

  for (int kb = 0; kb < nkb; ++kb) {
    const int jk = kb * 32;

    // ---- partial S = Q[16 x 128-slice] * K^T  (two 16x16 key tiles) ----
    v8f s0 = zero8(), s1 = zero8();
    #pragma unroll
    for (int ks = 0; ks < 4; ++ks) {
      const int kk = d0 + ks * 32;
      v16bf a = load_a_frag(Qrow + kk, lane);
      const bf16* kr0 =
          Kmat + ((size_t)b * TSEQ + (jk + row)) * DDIM + kk + g * 16;
      v16bf b0 = *(const v16bf*)kr0;
      v16bf b1 = *(const v16bf*)(kr0 + (size_t)16 * DDIM);
      s0 = wmma_bf16(a, b0, s0);
      s1 = wmma_bf16(a, b1, s1);
    }
    {
      const int M0 = g * 8;
      #pragma unroll
      for (int r = 0; r < 8; ++r) {
        S_part[wv][M0 + r][row]      = s0[r];
        S_part[wv][M0 + r][16 + row] = s1[r];
      }
    }
    __syncthreads();

    // ---- deterministic reduction across the 8 waves ----
    {
      const float* p0 = &S_part[0][0][0];
      float* sp = &S_lds[0][0];
      for (int i = threadIdx.x; i < 512; i += 256) {
        float s = 0.0f;
        #pragma unroll
        for (int w = 0; w < 8; ++w) s += p0[w * 512 + i];
        sp[i] = s;
      }
    }
    __syncthreads();

    // ---- online softmax for row `row` (lane pairs duplicate the work) ----
    const int qidx = q0 + row;
    float mb = -3.0e38f;
    #pragma unroll
    for (int j = 0; j < 32; ++j) {
      float s = ((jk + j) <= qidx) ? S_lds[row][j] * c : -3.0e38f;
      mb = fmaxf(mb, s);
    }
    const float m_new = fmaxf(m_run, mb);
    float l_new = 0.0f;
    bf16 pf[16];
    #pragma unroll
    for (int j = 0; j < 32; ++j) {
      const bool ok = (jk + j) <= qidx;
      float p = ok ? exp2f(S_lds[row][j] * c - m_new) : 0.0f;
      l_new += p;
      // scatter into A-fragment slots for this lane
      const int jh = j >> 4;
      if (jh == g)          pf[j & 15]      = (bf16)p;
    }
    const float fac = exp2f(m_run - m_new);
    l_run = l_run * fac + l_new;
    m_run = m_new;

    // ---- rescale accumulators (per-row factor via intra-wave LDS bcast) ----
    if (lane < 16) bc[wv][row] = fac;
    float fr[8];
    #pragma unroll
    for (int r = 0; r < 8; ++r) fr[r] = bc[wv][g * 8 + r];
    #pragma unroll
    for (int t = 0; t < 8; ++t)
      #pragma unroll
      for (int r = 0; r < 8; ++r) acc[t][r] *= fr[r];

    // ---- P * V over this wave's 128 output columns ----
    union { v16bf v; bf16 e[16]; } au;
    #pragma unroll
    for (int i = 0; i < 16; ++i) au.e[i] = pf[i];
    #pragma unroll
    for (int t = 0; t < 8; ++t) {
      const bf16* vr =
          Vt + ((size_t)b * DDIM + (d0 + t * 16 + row)) * TSEQ + jk + g * 16;
      v16bf bv = *(const v16bf*)vr;
      acc[t] = wmma_bf16(au.v, bv, acc[t]);
    }
    __syncthreads();   // protect S_part/S_lds before next iteration writes
  }

  // ---- finalize: divide by row sums, store fp32 ----
  if (lane < 16) bc[wv][row] = 1.0f / l_run;
  float ir[8];
  #pragma unroll
  for (int r = 0; r < 8; ++r) ir[r] = bc[wv][g * 8 + r];
  #pragma unroll
  for (int t = 0; t < 8; ++t) {
    const int col = d0 + t * 16 + row;
    #pragma unroll
    for (int r = 0; r < 8; ++r) {
      const int m = q0 + g * 8 + r;
      Out[((size_t)b * TSEQ + m) * DDIM + col] = acc[t][r] * ir[r];
    }
  }
}

// ---------------------------------------------------------------------------
extern "C" void kernel_launch(void* const* d_in, const int* in_sizes, int n_in,
                              void* d_out, int out_size, void* d_ws,
                              size_t ws_size, hipStream_t stream) {
  (void)in_sizes; (void)n_in; (void)out_size; (void)ws_size;
  const float* x  = (const float*)d_in[0];
  const float* wq = (const float*)d_in[1];
  const float* wk = (const float*)d_in[2];
  const float* wvp = (const float*)d_in[3];
  float* out = (float*)d_out;

  char* ws = (char*)d_ws;
  bf16* xb  = (bf16*)(ws);                       // 8M  elems (16 MB)
  bf16* wt  = (bf16*)(ws + ((size_t)16 << 20));  // 3M  elems ( 6 MB)
  bf16* Qb  = (bf16*)(ws + ((size_t)22 << 20));  // 8M  elems (16 MB)
  bf16* Kb  = (bf16*)(ws + ((size_t)38 << 20));  // 8M  elems (16 MB)
  bf16* Vt  = (bf16*)(ws + ((size_t)54 << 20));  // 8M  elems (16 MB) -> 70 MB

  const int NX = BATCH * TSEQ * DDIM;            // 8M
  cvt_bf16_kernel<<<NX / 256, 256, 0, stream>>>(x, xb, NX);
  const int NW = DDIM * DDIM;                    // 1M
  transpose_w_kernel<<<NW / 256, 256, 0, stream>>>(wq, wt);
  transpose_w_kernel<<<NW / 256, 256, 0, stream>>>(wk, wt + (size_t)NW);
  transpose_w_kernel<<<NW / 256, 256, 0, stream>>>(wvp, wt + (size_t)2 * NW);

  // M=8192 rows, 128 rows per block (8 waves x 16); N tiled 64 per wave.
  qkv_gemm_kernel<<<dim3(64, 16, 3), 256, 0, stream>>>(xb, wt, Qb, Kb, Vt);

  attn_kernel<<<dim3(TSEQ / 16, BATCH), 256, 0, stream>>>(Qb, Kb, Vt, out);
}
